// LERA_Module_66623532695848
// MI455X (gfx1250) — compile-verified, matrix operand order checked by reading, not run on hardware
//
#include <hip/hip_runtime.h>
#include <hip/hip_bf16.h>

// MI455X / gfx1250, wave32. Matrix ops via v_wmma_f32_16x16x32_bf16.
typedef __attribute__((ext_vector_type(16))) __bf16 v16bf;
typedef __attribute__((ext_vector_type(8)))  __bf16 v8bf;
typedef __attribute__((ext_vector_type(4)))  __bf16 v4bf;
typedef __attribute__((ext_vector_type(16))) float  v16f;
typedef __attribute__((ext_vector_type(8)))  float  v8f;
typedef __attribute__((ext_vector_type(4)))  float  v4f;

#define NHEAD 8
#define DHEAD 64
#define NB    8
#define CCH   512
#define NN    1024   // H*W

// 16-bit A/B fragment (ISA 7.12.2): per lane, elements 0..7 = K(8*half+0..7),
// elements 8..15 = K(16+8*half+0..7)  -> two contiguous 16-byte chunks.
__device__ inline v16bf load_frag_bf16(const __bf16* rowptr, int half) {
  v8bf lo = *(const v8bf*)(rowptr + 8 * half);
  v8bf hi = *(const v8bf*)(rowptr + 16 + 8 * half);
  return __builtin_shufflevector(lo, hi, 0, 1, 2, 3, 4, 5, 6, 7,
                                         8, 9, 10, 11, 12, 13, 14, 15);
}

// Same fragment pattern but sourced from fp32 memory (weights):
// 4x float4 b128 loads -> one vector convert (packed v_cvt bf16 ops).
__device__ inline v16bf load_frag_f32(const float* rowptr, int half) {
  v4f f0 = *(const v4f*)(rowptr + 8 * half);
  v4f f1 = *(const v4f*)(rowptr + 8 * half + 4);
  v4f f2 = *(const v4f*)(rowptr + 16 + 8 * half);
  v4f f3 = *(const v4f*)(rowptr + 16 + 8 * half + 4);
  v8f lo = __builtin_shufflevector(f0, f1, 0, 1, 2, 3, 4, 5, 6, 7);
  v8f hi = __builtin_shufflevector(f2, f3, 0, 1, 2, 3, 4, 5, 6, 7);
  v16f ff = __builtin_shufflevector(lo, hi, 0, 1, 2, 3, 4, 5, 6, 7,
                                            8, 9, 10, 11, 12, 13, 14, 15);
  return __builtin_convertvector(ff, v16bf);
}

// ---------------------------------------------------------------------------
// Kernel 1: Q/K projection.  Out[n, cout] = sum_c x[b][c][n] * W[cout][c] + bias
// One wave per 16x16 output tile. x (transposed operand) is staged through LDS
// with coalesced b128 loads; weights read as b128 directly. 16 bf16 WMMAs.
// Q is pre-scaled by 1/sqrt(d)=0.125. Q/K stored bf16 as [b][head][n][d].
// ---------------------------------------------------------------------------
__global__ __launch_bounds__(32) void proj_qk(const float* __restrict__ x,
                                              const float* __restrict__ W,
                                              const float* __restrict__ bias,
                                              __bf16* __restrict__ Qbf,
                                              __bf16* __restrict__ Kbf) {
  __shared__ __attribute__((aligned(16))) __bf16 xt[16][32]; // [n][c] bf16 tile

  int bid   = blockIdx.x;
  int ct    = bid & 63;            // 64 cout tiles (32 for Q, 32 for K)
  int ntile = (bid >> 6) & 63;     // 64 n tiles
  int b     = bid >> 12;
  int lane  = threadIdx.x & 31;
  int half  = lane >> 4, row = lane & 15;
  bool isQ  = ct < 32;
  int cq    = isQ ? ct : (ct - 32);
  int crow0 = (isQ ? 0 : CCH) + cq * 16;   // row base into in_proj_weight
  int n0    = ntile * 16;
  const float* xb   = x + (size_t)b * CCH * NN;
  const float* Wrow = W + (size_t)(crow0 + row) * CCH;

  v8f acc = {};
  for (int kcb = 0; kcb < CCH; kcb += 32) {
    // Stage x tile: lane owns channel row c = kcb+lane, 16 contiguous n values.
    const float* xr = xb + (size_t)(kcb + lane) * NN + n0;
    v4bf b0 = __builtin_convertvector(*(const v4f*)(xr + 0),  v4bf);
    v4bf b1 = __builtin_convertvector(*(const v4f*)(xr + 4),  v4bf);
    v4bf b2 = __builtin_convertvector(*(const v4f*)(xr + 8),  v4bf);
    v4bf b3 = __builtin_convertvector(*(const v4f*)(xr + 12), v4bf);
    if (kcb + 32 < CCH) {
      __builtin_prefetch(xb + (size_t)(kcb + 32 + lane) * NN + n0, 0, 1);
      __builtin_prefetch(Wrow + kcb + 32, 0, 1);
    }
#pragma unroll
    for (int n = 0; n < 4; ++n) {
      xt[n + 0][lane]  = b0[n];
      xt[n + 4][lane]  = b1[n];
      xt[n + 8][lane]  = b2[n];
      xt[n + 12][lane] = b3[n];
    }
    __syncthreads();

    v16bf a  = load_frag_bf16(&xt[row][0], half);   // A[m=n][k=c] via ds_load_b128
    v16bf bb = load_frag_f32(Wrow + kcb, half);     // B[k][n=cout] = W[cout][k]
    acc = __builtin_amdgcn_wmma_f32_16x16x32_bf16(false, a, false, bb,
                                                  (short)0, acc, false, false);
    __syncthreads();
  }

  int head = (cq * 16) / DHEAD;
  int dd0  = (cq * 16) % DHEAD;
  __bf16* dst = (isQ ? Qbf : Kbf) + (size_t)(b * NHEAD + head) * NN * DHEAD;
#pragma unroll
  for (int v = 0; v < 8; ++v) {
    int m = v + 8 * half;        // D layout: lanes 0-15 -> M=v, lanes 16-31 -> M=v+8
    int n = row;
    float val = acc[v] + bias[crow0 + n];
    if (isQ) val *= 0.125f;      // fold 1/sqrt(d) into Q
    dst[(size_t)(n0 + m) * DHEAD + dd0 + n] = (__bf16)val;
  }
}

// ---------------------------------------------------------------------------
// Kernel 2: fused scores + softmax + head-mean + second softmax + q-mean.
// One block per (b, 16-query tile). Scores never leave registers/LDS.
// Each of 8 waves owns 8 key tiles (128 keys); per head: 16 bf16 WMMAs.
// Fragments assembled from two global_load_b128 per half (row-major bf16).
// ---------------------------------------------------------------------------
__global__ __launch_bounds__(256) void attn_reduce(const __bf16* __restrict__ Qbf,
                                                   const __bf16* __restrict__ Kbf,
                                                   float* __restrict__ ks) {
  extern __shared__ float smem[];
  float* accs  = smem;               // 16*1024  mean-over-heads attention rows
  float* wred  = accs + 16 * 1024;   // 8*16     per-wave row partials
  float* rowmx = wred + 8 * 16;      // 16
  float* rowsm = rowmx + 16;         // 16
  float* red2  = rowsm + 16;         // 16*16    second-softmax scratch

  int tid  = threadIdx.x;
  int lane = tid & 31, wave = tid >> 5;
  int half = lane >> 4, row = lane & 15;
  int b  = blockIdx.x >> 6;
  int q0 = (blockIdx.x & 63) * 16;

  for (int i = tid; i < 16 * 1024; i += 256) accs[i] = 0.0f;
  __syncthreads();

  for (int h = 0; h < NHEAD; ++h) {
    const __bf16* Qh = Qbf + (size_t)(b * NHEAD + h) * NN * DHEAD;
    const __bf16* Kh = Kbf + (size_t)(b * NHEAD + h) * NN * DHEAD;
    if (h + 1 < NHEAD) {   // prefetch next head's Q rows (global_prefetch_b8)
      __builtin_prefetch(Qh + (size_t)NN * DHEAD + (size_t)(q0 + row) * DHEAD, 0, 1);
    }

    v16bf aq[2];
#pragma unroll
    for (int s = 0; s < 2; ++s)
      aq[s] = load_frag_bf16(Qh + (size_t)(q0 + row) * DHEAD + s * 32, half);

    v8f cf[8];
#pragma unroll
    for (int tt = 0; tt < 8; ++tt) {
      int k0 = (wave + tt * 8) * 16;
      const __bf16* Krow = Kh + (size_t)(k0 + row) * DHEAD;
      v8f c = {};
#pragma unroll
      for (int s = 0; s < 2; ++s) {
        v16bf bb = load_frag_bf16(Krow + s * 32, half); // B[kd][n=key]=K[k0+n][kd]
        c = __builtin_amdgcn_wmma_f32_16x16x32_bf16(false, aq[s], false, bb,
                                                    (short)0, c, false, false);
      }
      cf[tt] = c;
    }

    // ---- per-row max: registers -> shfl_xor within 16-lane half -> LDS across waves
    float lred[8];
#pragma unroll
    for (int v = 0; v < 8; ++v) {
      float m = -1e30f;
#pragma unroll
      for (int tt = 0; tt < 8; ++tt) m = fmaxf(m, cf[tt][v]);
#pragma unroll
      for (int off = 1; off < 16; off <<= 1) m = fmaxf(m, __shfl_xor(m, off, 32));
      lred[v] = m;
    }
    if (row == 0) {
#pragma unroll
      for (int v = 0; v < 8; ++v) wred[wave * 16 + v + 8 * half] = lred[v];
    }
    __syncthreads();
    if (tid < 16) {
      float m = -1e30f;
      for (int w = 0; w < 8; ++w) m = fmaxf(m, wred[w * 16 + tid]);
      rowmx[tid] = m;
    }
    __syncthreads();

    // ---- exp in-place + per-row sum
#pragma unroll
    for (int v = 0; v < 8; ++v) {
      int m = v + 8 * half;
      float rm = rowmx[m];
      float s = 0.0f;
#pragma unroll
      for (int tt = 0; tt < 8; ++tt) {
        float e = __expf(cf[tt][v] - rm);
        cf[tt][v] = e;
        s += e;
      }
#pragma unroll
      for (int off = 1; off < 16; off <<= 1) s += __shfl_xor(s, off, 32);
      lred[v] = s;
    }
    if (row == 0) {
#pragma unroll
      for (int v = 0; v < 8; ++v) wred[wave * 16 + v + 8 * half] = lred[v];
    }
    __syncthreads();
    if (tid < 16) {
      float s = 0.0f;
      for (int w = 0; w < 8; ++w) s += wred[w * 16 + tid];
      rowsm[tid] = s;
    }
    __syncthreads();

    // ---- accumulate head-mean attention row (each (m,col) touched by one lane)
#pragma unroll
    for (int v = 0; v < 8; ++v) {
      int m = v + 8 * half;
      float inv = 0.125f / rowsm[m];
#pragma unroll
      for (int tt = 0; tt < 8; ++tt) {
        int col = (wave + tt * 8) * 16 + row;
        accs[m * 1024 + col] += cf[tt][v] * inv;
      }
    }
    __syncthreads();
  }

  // ---- second softmax over accs/sqrt(N) (=/32), then mean over queries -> ks
  {
    int r = tid >> 4, c16 = tid & 15;
    float m = -1e30f;
    for (int c = c16; c < 1024; c += 16) m = fmaxf(m, accs[r * 1024 + c]);
    red2[r * 16 + c16] = m;
    __syncthreads();
    if (tid < 16) {
      float mm = -1e30f;
      for (int j = 0; j < 16; ++j) mm = fmaxf(mm, red2[tid * 16 + j]);
      rowmx[tid] = mm * 0.03125f;
    }
    __syncthreads();
    float rm = rowmx[r];
    float s = 0.0f;
    for (int c = c16; c < 1024; c += 16)
      s += __expf(accs[r * 1024 + c] * 0.03125f - rm);
    red2[r * 16 + c16] = s;
    __syncthreads();
    if (tid < 16) {
      float ss = 0.0f;
      for (int j = 0; j < 16; ++j) ss += red2[tid * 16 + j];
      rowsm[tid] = ss;
    }
    __syncthreads();
    for (int c = tid; c < 1024; c += 256) {
      float s2 = 0.0f;
#pragma unroll
      for (int r2 = 0; r2 < 16; ++r2)
        s2 += __expf(accs[r2 * 1024 + c] * 0.03125f - rowmx[r2]) / rowsm[r2];
      atomicAdd(&ks[b * 1024 + c], s2 * (1.0f / 1024.0f));
    }
  }
}

// ---------------------------------------------------------------------------
// Kernel 3: top-8 of key_scores per batch, gather x rows, max-pool -> mp[b][c]
// ---------------------------------------------------------------------------
__global__ __launch_bounds__(256) void topk_pool(const float* __restrict__ ks,
                                                 const float* __restrict__ x,
                                                 float* __restrict__ mp) {
  __shared__ float sks[1024];
  __shared__ int   sidx[8];
  __shared__ float rmax[256];
  __shared__ int   ridx[256];
  int b = blockIdx.x, tid = threadIdx.x;
  for (int c = tid; c < 1024; c += 256) sks[c] = ks[b * 1024 + c];
  __syncthreads();
  for (int j = 0; j < 8; ++j) {
    float best = -1e30f; int bi = 0x7FFFFFFF;
    for (int c = tid; c < 1024; c += 256) {
      float v = sks[c];
      if (v > best || (v == best && c < bi)) { best = v; bi = c; }
    }
    rmax[tid] = best; ridx[tid] = bi;
    __syncthreads();
    for (int s = 128; s > 0; s >>= 1) {
      if (tid < s) {
        float v = rmax[tid + s]; int i2 = ridx[tid + s];
        if (v > rmax[tid] || (v == rmax[tid] && i2 < ridx[tid])) {
          rmax[tid] = v; ridx[tid] = i2;
        }
      }
      __syncthreads();
    }
    if (tid == 0) { sidx[j] = ridx[0]; sks[ridx[0]] = -1e30f; }
    __syncthreads();
  }
  for (int c = tid; c < CCH; c += 256) {
    float m = -1e30f;
#pragma unroll
    for (int j = 0; j < 8; ++j)
      m = fmaxf(m, x[((size_t)b * CCH + c) * NN + sidx[j]]);
    mp[b * CCH + c] = m;
  }
}

// ---------------------------------------------------------------------------
// Kernel 4: out[b][c][n] = x[b][c][n] + mp[b][c]
// ---------------------------------------------------------------------------
__global__ void final_add(const float* __restrict__ x, const float* __restrict__ mp,
                          float* __restrict__ out, int total) {
  int i = blockIdx.x * 256 + threadIdx.x;
  if (i < total) out[i] = x[i] + mp[i >> 10];   // N = 1024
}

__global__ void init_ks(float* ks, int n) {
  int i = blockIdx.x * 256 + threadIdx.x;
  if (i < n) ks[i] = 0.0f;
}

extern "C" void kernel_launch(void* const* d_in, const int* in_sizes, int n_in,
                              void* d_out, int out_size, void* d_ws, size_t ws_size,
                              hipStream_t stream) {
  const float* x    = (const float*)d_in[0];   // (8,512,32,32)
  const float* W    = (const float*)d_in[1];   // (1536,512)
  const float* bias = (const float*)d_in[2];   // (1536,)
  float* out = (float*)d_out;

  char* ws = (char*)d_ws;
  const size_t qkBytes = (size_t)NB * NHEAD * NN * DHEAD * sizeof(unsigned short); // 8.39MB
  __bf16* Qbf = (__bf16*)ws;
  __bf16* Kbf = (__bf16*)(ws + qkBytes);
  float*  ks  = (float*)(ws + 2 * qkBytes);
  float*  mp  = (float*)(ws + 2 * qkBytes + NB * NN * sizeof(float));

  init_ks<<<(NB * NN + 255) / 256, 256, 0, stream>>>(ks, NB * NN);

  // B(8) * n-tiles(64) * cout-tiles(Q:32 + K:32) one-wave blocks
  proj_qk<<<NB * 64 * 64, 32, 0, stream>>>(x, W, bias, Qbf, Kbf);

  size_t smem = (size_t)(16 * 1024 + 8 * 16 + 16 + 16 + 16 * 16) * sizeof(float); // ~67KB (dynamic, CDNA5 320KB/WGP)
  attn_reduce<<<NB * 64, 256, smem, stream>>>(Qbf, Kbf, ks);

  topk_pool<<<NB, 256, 0, stream>>>(ks, x, mp);

  int total = NB * CCH * NN;
  final_add<<<total / 256, 256, 0, stream>>>(x, mp, out, total);
}
